// FastSpeech2ConformerAttention_75995151335633
// MI455X (gfx1250) — matrix operand order repaired and checked by reading, not compiled, not run on hardware
//
#include <hip/hip_runtime.h>

// FastSpeech2 Conformer relative-position attention for MI455X (gfx1250).
// bf16 WMMA (v_wmma_f32_16x16x32_bf16) everywhere, f32 accumulate,
// flash-attention style fusion so [t,t] / [t,2t-1] scores never touch HBM.
// Round 3: __builtin_amdgcn_sched_barrier(0) between batched fragment loads
// and WMMA groups so the scheduler cannot re-serialize them (progressive
// s_wait_loadcnt instead of wait-0 before every matrix op); both rel-pos
// G subtiles share one LDS phase (half the barriers per 32-wide k strip).

#define HID   384
#define HD    96
#define NHEAD 4
#define BATCH 8
#define TSEQ  2048
#define NPOS  (2 * TSEQ - 1) /* 4095 */

typedef __bf16 bf16_t;
typedef __attribute__((ext_vector_type(16))) __bf16 v16bf;
typedef __attribute__((ext_vector_type(8)))  __bf16 v8bf;
typedef __attribute__((ext_vector_type(8)))  float  v8f;

#if defined(__has_builtin) && __has_builtin(__builtin_amdgcn_sched_barrier)
#define SCHED_FENCE() __builtin_amdgcn_sched_barrier(0)
#else
#define SCHED_FENCE()
#endif

static __device__ __forceinline__ v8f zero8() {
    v8f z;
#pragma unroll
    for (int i = 0; i < 8; ++i) z[i] = 0.0f;
    return z;
}

static __device__ __forceinline__ v8f wmma_bf16(v16bf a, v16bf b, v8f c) {
    // (neg_a, A, neg_b, B, c_mod, C, reuse_a, reuse_b)
    return __builtin_amdgcn_wmma_f32_16x16x32_bf16(false, a, false, b, (short)0, c,
                                                   false, false);
}

// A-operand 16x32 bf16 fragment from row-major [row, k] matrix.
// ISA 7.12.2: lane&15 = M, h = lane>>4; elem e<8 -> K=8h+e ; e>=8 -> K=16+8h+(e-8).
static __device__ __forceinline__ v16bf load_a(const bf16_t* base, long row0,
                                               long maxrow, int ld, int lane) {
    int m = lane & 15, h = lane >> 4;
    long r = row0 + m;
    r = r < 0 ? 0 : (r > maxrow ? maxrow : r);
    const bf16_t* p = base + r * (long)ld + 8 * h;
    v8bf c0 = *(const v8bf*)(p);
    v8bf c1 = *(const v8bf*)(p + 16);
    v16bf f;
#pragma unroll
    for (int i = 0; i < 8; ++i) { f[i] = c0[i]; f[i + 8] = c1[i]; }
    return f;
}

// B-operand 32x16 bf16 fragment; B[k,n] = base[(row0+n)*ld + k].
// dense-B: lane&15 = N, h = lane>>4; elem e<8 -> K=8h+e ; e>=8 -> K=16+8h+(e-8).
static __device__ __forceinline__ v16bf load_b(const bf16_t* base, long row0,
                                               long maxrow, int ld, int lane) {
    int n = lane & 15, h = lane >> 4;
    long r = row0 + n;
    r = r < 0 ? 0 : (r > maxrow ? maxrow : r);
    const bf16_t* p = base + r * (long)ld + 8 * h;
    v8bf c0 = *(const v8bf*)(p);
    v8bf c1 = *(const v8bf*)(p + 16);
    v16bf f;
#pragma unroll
    for (int i = 0; i < 8; ++i) { f[i] = c0[i]; f[i + 8] = c1[i]; }
    return f;
}

// ---------------- conversion kernels ----------------
__global__ void k_cvt(const float* __restrict__ s, bf16_t* __restrict__ d, int n) {
    int i = blockIdx.x * blockDim.x + threadIdx.x;
    if (i < n) d[i] = (bf16_t)s[i];
}

// Wt[n*384 + k] = W[k*384 + n]  (so B-fragments read contiguous k)
__global__ void k_cvt_T(const float* __restrict__ W, bf16_t* __restrict__ Wt) {
    int i = blockIdx.x * blockDim.x + threadIdx.x;
    if (i < HID * HID) {
        int n = i / HID, k = i % HID;
        Wt[i] = (bf16_t)W[k * HID + n];
    }
}

// ---------------- projection GEMM: Y = X(bf16)[R,384] @ Wt^T + b ----------------
// mode 0: Q -> writes Qu = (y + pos_bias_u)/sqrt(96), Qv = (y + pos_bias_v)/sqrt(96)
//         into [b,h,t,d]  (softmax scale pre-folded)
// mode 1: K -> [b,h,t,d]        mode 2: V -> transposed [b,h,d,t]
// mode 3: Pos -> [h, 4095, d]   (no bias)
__global__ void __launch_bounds__(32)
k_proj(const bf16_t* __restrict__ X, const bf16_t* __restrict__ Wt,
       const float* __restrict__ bias, const float* __restrict__ ub,
       const float* __restrict__ vb, bf16_t* __restrict__ o0,
       bf16_t* __restrict__ o1, int R, int mode) {
    int lane = threadIdx.x;
    long row0 = (long)blockIdx.x * 16;
    int h = blockIdx.y;          // head = 96-wide output chunk
    int col0 = h * HD;

    v8f acc[6];
#pragma unroll
    for (int i = 0; i < 6; ++i) acc[i] = zero8();

    for (int kk = 0; kk < HID; kk += 32) {
        // all fragment loads issue first; fence keeps WMMAs below them
        v16bf a = load_a(X + kk, row0, (long)R - 1, HID, lane);
        v16bf bfr[6];
#pragma unroll
        for (int nt = 0; nt < 6; ++nt)
            bfr[nt] = load_b(Wt + kk, col0 + nt * 16, HID - 1, HID, lane);
        SCHED_FENCE();
#pragma unroll
        for (int nt = 0; nt < 6; ++nt)
            acc[nt] = wmma_bf16(a, bfr[nt], acc[nt]);
        SCHED_FENCE();
    }

    const float rscale = 0.1020620726f; // 1/sqrt(96), folded into Q
    int M0 = (lane >> 4) * 8, nc = lane & 15;
#pragma unroll
    for (int nt = 0; nt < 6; ++nt) {
        int d = nt * 16 + nc;            // 0..95 within head
        int col = col0 + d;              // 0..383
        float bs = bias ? bias[col] : 0.0f;
#pragma unroll
        for (int r = 0; r < 8; ++r) {
            long row = row0 + M0 + r;
            if (row >= R) continue;
            float y = acc[nt][r] + bs;
            if (mode == 0) {
                long bb = row / TSEQ, t = row % TSEQ;
                long o = (((bb * NHEAD + h) * TSEQ) + t) * HD + d;
                o0[o] = (bf16_t)((y + ub[h * HD + d]) * rscale);
                o1[o] = (bf16_t)((y + vb[h * HD + d]) * rscale);
            } else if (mode == 1) {
                long bb = row / TSEQ, t = row % TSEQ;
                o0[(((bb * NHEAD + h) * TSEQ) + t) * HD + d] = (bf16_t)y;
            } else if (mode == 2) {
                long bb = row / TSEQ, t = row % TSEQ;
                o0[(((bb * NHEAD + h) * HD) + d) * (long)TSEQ + t] = (bf16_t)y;
            } else { // pos
                o0[((long)h * NPOS + row) * HD + d] = (bf16_t)y;
            }
        }
    }
}

// ---------------- fused rel-pos flash attention ----------------
// grid (T/16, H, B), 1 wave per block. Per 16-row q-tile: loop 32-wide k strips,
// S = Qu K^T + shift(Qv P^T) via windowed G-tile + LDS diagonal gather,
// online softmax, O += P V  (V pre-transposed). Scale pre-folded into Qu/Qv.
__global__ void __launch_bounds__(32)
k_attn(const bf16_t* __restrict__ Qu, const bf16_t* __restrict__ Qv,
       const bf16_t* __restrict__ Km, const bf16_t* __restrict__ Vt,
       const bf16_t* __restrict__ P, const int* __restrict__ amask,
       bf16_t* __restrict__ ctx) {
    __shared__ alignas(16) float  ldsG[16 * 64];  // both subtiles' G bands
    __shared__ alignas(16) bf16_t ldsP[16 * 32];

    int lane = threadIdx.x;
    int q0 = blockIdx.x * 16;
    int h = blockIdx.y, b = blockIdx.z;
    long off = ((long)(b * NHEAD + h)) * TSEQ * HD;
    const bf16_t* Qu_p = Qu + off + (long)q0 * HD;
    const bf16_t* Qv_p = Qv + off + (long)q0 * HD;
    const bf16_t* K_p  = Km + off;                 // [t, d]
    const bf16_t* Vt_p = Vt + off;                 // [d, t]
    const bf16_t* P_p  = P + (long)h * NPOS * HD;  // [n, d]

    v16bf qu[3], qv[3];
#pragma unroll
    for (int kc = 0; kc < 3; ++kc) {
        qu[kc] = load_a(Qu_p + kc * 32, 0, 15, HD, lane);
        qv[kc] = load_a(Qv_p + kc * 32, 0, 15, HD, lane);
    }

    v8f accO[6];
#pragma unroll
    for (int i = 0; i < 6; ++i) accO[i] = zero8();
    float m8[8], l8[8];
#pragma unroll
    for (int r = 0; r < 8; ++r) { m8[r] = -3.0e38f; l8[r] = 0.0f; }

    int M0 = (lane >> 4) * 8, nc = lane & 15;

    for (int k0 = 0; k0 < TSEQ; k0 += 32) {
        if (k0 + 32 < TSEQ) { // hint next strip into cache (global_prefetch_b8)
            __builtin_prefetch(K_p + (long)(k0 + 32) * HD, 0, 1);
            __builtin_prefetch(Vt_p + (long)(k0 + 32), 0, 1);
        }
        v8f accS[2];
        __syncthreads();      // ldsG gathers of previous strip are done
#pragma unroll
        for (int j = 0; j < 2; ++j) {
            int n0 = k0 + j * 16;
            long w0 = (long)(TSEQ - 1) + n0 - q0 - 15;
            // batch all 9 score-side fragment loads before any WMMA
            v16bf fr[9];
#pragma unroll
            for (int kc = 0; kc < 3; ++kc) {
                fr[kc]          = load_b(K_p + kc * 32, n0, TSEQ - 1, HD, lane);
                fr[3 + 2 * kc]  = load_b(P_p + kc * 32, w0, NPOS - 1, HD, lane);
                fr[4 + 2 * kc]  = load_b(P_p + kc * 32, w0 + 16, NPOS - 1, HD, lane);
            }
            SCHED_FENCE();
            // content scores: Qu . K^T   (Qu pre-scaled by 1/sqrt(96))
            accS[j] = zero8();
#pragma unroll
            for (int kc = 0; kc < 3; ++kc)
                accS[j] = wmma_bf16(qu[kc], fr[kc], accS[j]);
            // position band: G = Qv . P[w0 .. w0+31]^T  (rows stay in [0,4094],
            // one clamped edge row is never read back)
            v8f g0 = zero8(), g1 = zero8();
#pragma unroll
            for (int kc = 0; kc < 3; ++kc) {
                g0 = wmma_bf16(qv[kc], fr[3 + 2 * kc], g0);
                g1 = wmma_bf16(qv[kc], fr[4 + 2 * kc], g1);
            }
            SCHED_FENCE();
#pragma unroll
            for (int r = 0; r < 8; ++r) {
                ldsG[(M0 + r) * 64 + j * 32 + nc]      = g0[r];
                ldsG[(M0 + r) * 64 + j * 32 + 16 + nc] = g1[r];
            }
        }
        __syncthreads();      // G bands visible; single sync for both subtiles
        float sv[2][8], va[2];
#pragma unroll
        for (int j = 0; j < 2; ++j) {
            int mv = amask[b * TSEQ + k0 + j * 16 + nc];
            va[j] = mv ? 1.0f : 0.0f;
#pragma unroll
            for (int r = 0; r < 8; ++r) {
                int Mm = M0 + r;
                // bd[q,k] = G[dq, 15 + dk - dq]  (rel-shift diagonal gather)
                float bd = ldsG[Mm * 64 + j * 32 + (nc - Mm + 15)];
                float s = accS[j][r] + bd;  // already scaled via Q
                sv[j][r] = mv ? s : -3.0e38f;
            }
        }
        // online softmax over the 32-wide strip (row lives in one 16-lane half)
        float pv[2][8];
#pragma unroll
        for (int r = 0; r < 8; ++r) {
            float tm = fmaxf(sv[0][r], sv[1][r]);
#pragma unroll
            for (int o = 1; o < 16; o <<= 1) tm = fmaxf(tm, __shfl_xor(tm, o, 32));
            float mn = fmaxf(m8[r], tm);
            float corr = __expf(m8[r] - mn);
            float p0 = __expf(sv[0][r] - mn) * va[0];
            float p1 = __expf(sv[1][r] - mn) * va[1];
            float rs = p0 + p1;
#pragma unroll
            for (int o = 1; o < 16; o <<= 1) rs += __shfl_xor(rs, o, 32);
            l8[r] = l8[r] * corr + rs;
            m8[r] = mn;
            pv[0][r] = p0; pv[1][r] = p1;
#pragma unroll
            for (int dt = 0; dt < 6; ++dt) accO[dt][r] *= corr;
        }
        // probabilities -> bf16 A-fragment via LDS transpose
        __syncthreads();
#pragma unroll
        for (int r = 0; r < 8; ++r) {
            ldsP[(M0 + r) * 32 + nc]      = (bf16_t)pv[0][r];
            ldsP[(M0 + r) * 32 + 16 + nc] = (bf16_t)pv[1][r];
        }
        __syncthreads();
        v16bf pf = load_a(ldsP, 0, 15, 32, lane);
        // batch V fragments before the PV WMMAs
        v16bf vf[6];
#pragma unroll
        for (int dt = 0; dt < 6; ++dt)
            vf[dt] = load_b(Vt_p + k0, dt * 16, HD - 1, TSEQ, lane);
        SCHED_FENCE();
#pragma unroll
        for (int dt = 0; dt < 6; ++dt)
            accO[dt] = wmma_bf16(pf, vf[dt], accO[dt]);
        SCHED_FENCE();
    }
    // finalize: ctx[b, t, h*96 + d] bf16
#pragma unroll
    for (int r = 0; r < 8; ++r) {
        float inv = 1.0f / fmaxf(l8[r], 1e-30f);
        long t = q0 + M0 + r;
#pragma unroll
        for (int dt = 0; dt < 6; ++dt) {
            long o = ((long)b * TSEQ + t) * HID + h * HD + dt * 16 + nc;
            ctx[o] = (bf16_t)(accO[dt][r] * inv);
        }
    }
}

// ---------------- output projection: out(f32) = ctx @ Wo + bo ----------------
__global__ void __launch_bounds__(32)
k_outproj(const bf16_t* __restrict__ X, const bf16_t* __restrict__ Wt,
          const float* __restrict__ bias, float* __restrict__ out, int R) {
    int lane = threadIdx.x;
    long row0 = (long)blockIdx.x * 16;
    int col0 = blockIdx.y * HD;
    v8f acc[6];
#pragma unroll
    for (int i = 0; i < 6; ++i) acc[i] = zero8();
    for (int kk = 0; kk < HID; kk += 32) {
        v16bf a = load_a(X + kk, row0, (long)R - 1, HID, lane);
        v16bf bfr[6];
#pragma unroll
        for (int nt = 0; nt < 6; ++nt)
            bfr[nt] = load_b(Wt + kk, col0 + nt * 16, HID - 1, HID, lane);
        SCHED_FENCE();
#pragma unroll
        for (int nt = 0; nt < 6; ++nt)
            acc[nt] = wmma_bf16(a, bfr[nt], acc[nt]);
        SCHED_FENCE();
    }
    int M0 = (lane >> 4) * 8, nc = lane & 15;
#pragma unroll
    for (int nt = 0; nt < 6; ++nt) {
        int col = col0 + nt * 16 + nc;
        float bs = bias[col];
#pragma unroll
        for (int r = 0; r < 8; ++r) {
            long row = row0 + M0 + r;
            if (row < R) out[row * HID + col] = acc[nt][r] + bs;
        }
    }
}

extern "C" void kernel_launch(void* const* d_in, const int* in_sizes, int n_in,
                              void* d_out, int out_size, void* d_ws, size_t ws_size,
                              hipStream_t stream) {
    const float* hs    = (const float*)d_in[0];
    const float* pos   = (const float*)d_in[1];
    const int*   amask = (const int*)d_in[2];
    const float* Wq = (const float*)d_in[3];
    const float* bq = (const float*)d_in[4];
    const float* Wk = (const float*)d_in[5];
    const float* bk = (const float*)d_in[6];
    const float* Wv = (const float*)d_in[7];
    const float* bv = (const float*)d_in[8];
    const float* Wo = (const float*)d_in[9];
    const float* bo = (const float*)d_in[10];
    const float* Wp = (const float*)d_in[11];
    const float* pbu = (const float*)d_in[12];
    const float* pbv = (const float*)d_in[13];
    float* out = (float*)d_out;

    const long R = (long)BATCH * TSEQ;          // 16384
    char* ws = (char*)d_ws;
    size_t off = 0;
    auto alloc = [&](size_t bytes) -> void* {
        void* p = ws + off;
        off += (bytes + 255) & ~(size_t)255;
        return p;
    };
    bf16_t* hsb  = (bf16_t*)alloc(R * HID * 2);
    bf16_t* posb = (bf16_t*)alloc((size_t)NPOS * HID * 2);
    bf16_t* WqT = (bf16_t*)alloc(HID * HID * 2);
    bf16_t* WkT = (bf16_t*)alloc(HID * HID * 2);
    bf16_t* WvT = (bf16_t*)alloc(HID * HID * 2);
    bf16_t* WoT = (bf16_t*)alloc(HID * HID * 2);
    bf16_t* WpT = (bf16_t*)alloc(HID * HID * 2);
    bf16_t* Qub = (bf16_t*)alloc(R * HID * 2);
    bf16_t* Qvb = (bf16_t*)alloc(R * HID * 2);
    bf16_t* Kb  = (bf16_t*)alloc(R * HID * 2);
    bf16_t* Vtb = (bf16_t*)alloc(R * HID * 2);
    bf16_t* Pb  = (bf16_t*)alloc((size_t)NHEAD * NPOS * HD * 2);
    bf16_t* ctx = (bf16_t*)alloc(R * HID * 2);
    (void)ws_size; (void)in_sizes; (void)n_in; (void)out_size;

    // 0) f32 -> bf16 conversions (+ weight transposes)
    int nh = (int)(R * HID);
    k_cvt<<<(nh + 255) / 256, 256, 0, stream>>>(hs, hsb, nh);
    int np = NPOS * HID;
    k_cvt<<<(np + 255) / 256, 256, 0, stream>>>(pos, posb, np);
    int nw = HID * HID;
    int gw = (nw + 255) / 256;
    k_cvt_T<<<gw, 256, 0, stream>>>(Wq, WqT);
    k_cvt_T<<<gw, 256, 0, stream>>>(Wk, WkT);
    k_cvt_T<<<gw, 256, 0, stream>>>(Wv, WvT);
    k_cvt_T<<<gw, 256, 0, stream>>>(Wo, WoT);
    k_cvt_T<<<gw, 256, 0, stream>>>(Wp, WpT);

    // 1) projections (WMMA GEMMs)
    dim3 gp((unsigned)(R / 16), NHEAD);
    k_proj<<<gp, 32, 0, stream>>>(hsb, WqT, bq, pbu, pbv, Qub, Qvb, (int)R, 0);
    k_proj<<<gp, 32, 0, stream>>>(hsb, WkT, bk, nullptr, nullptr, Kb, nullptr, (int)R, 1);
    k_proj<<<gp, 32, 0, stream>>>(hsb, WvT, bv, nullptr, nullptr, Vtb, nullptr, (int)R, 2);
    dim3 gpp((NPOS + 15) / 16, NHEAD);
    k_proj<<<gpp, 32, 0, stream>>>(posb, WpT, nullptr, nullptr, nullptr, Pb, nullptr, NPOS, 3);

    // 2) fused rel-pos flash attention
    dim3 ga(TSEQ / 16, NHEAD, BATCH);
    k_attn<<<ga, 32, 0, stream>>>(Qub, Qvb, Kb, Vtb, Pb, amask, ctx);

    // 3) output projection to f32
    dim3 go((unsigned)(R / 16), NHEAD);
    k_outproj<<<go, 32, 0, stream>>>(ctx, WoT, bo, out, (int)R);
}
